// CondLSTMProposal_18227841204489
// MI455X (gfx1250) — matrix operand order repaired and checked by reading, not compiled, not run on hardware
//
#include <hip/hip_runtime.h>
#include <cstdint>
#include <cstddef>

#define BDIM 2048
#define DDIM 256
#define KDIM 20
#define UDIM 512
#define EDIM 64
#define HDIM 256
#define HIDD 512
#define G4H  1024
#define LN_EPS 1e-5f

typedef __bf16 v16bf __attribute__((ext_vector_type(16)));
typedef float  v8f   __attribute__((ext_vector_type(8)));

union FragBF { v16bf v; uint4 q[2]; unsigned short u[16]; };

static __device__ __forceinline__ unsigned short f2bf(float f) {
  unsigned u = __float_as_uint(f);
  u += 0x7FFFu + ((u >> 16) & 1u);
  return (unsigned short)(u >> 16);
}
static __device__ __forceinline__ float bf2f(unsigned short s) {
  return __uint_as_float(((unsigned)s) << 16);
}
static __device__ __forceinline__ float sigmoidf_(float x) {
  return 1.0f / (1.0f + expf(-x));
}

// ---- CDNA5 async global->LDS copy (ASYNCcnt path), per CDNA5 ISA ----
// VDST = LDS byte address (VGPR), VADDR = 64-bit global address (VGPR pair), SADDR = off.
static __device__ __forceinline__ void async_copy16(unsigned ldsoff, const void* gptr) {
  asm volatile("global_load_async_to_lds_b128 %0, %1, off"
               :: "v"(ldsoff), "v"((unsigned long long)gptr)
               : "memory");
}
static __device__ __forceinline__ void async_wait0() {
  asm volatile("s_wait_asynccnt 0" ::: "memory");
}
template <typename T>
static __device__ __forceinline__ unsigned lds_off(T* p) {
  // generic pointers to LDS carry the wave-relative offset in the low 32 bits
  return (unsigned)(unsigned long long)p;
}

// A fragment (16x32 bf16, M rows of src): lane m=lane&15, k = k0 + {0,8}+{0..7} and +16.
static __device__ __forceinline__ v16bf load_a_frag(const unsigned short* base, int row0,
                                                    int ld, int k0, int lane) {
  int m  = lane & 15;
  int kb = (lane >> 4) << 3;  // 0 or 8
  const unsigned short* p = base + (size_t)(row0 + m) * ld + k0 + kb;
  FragBF f;
  f.q[0] = *(const uint4*)(p);
  f.q[1] = *(const uint4*)(p + 16);
  return f.v;
}
// B fragment (32x16 bf16) from row-major weight W[N,K]: B[k][n] = W[n][k].
static __device__ __forceinline__ v16bf load_b_frag(const unsigned short* w, int ld,
                                                    int n0, int k0, int lane) {
  int n  = lane & 15;
  int kh = (lane >> 4) << 3;  // 0 or 8
  const unsigned short* p = w + (size_t)(n0 + n) * ld + k0 + kh;
  FragBF f;
  f.q[0] = *(const uint4*)(p);
  f.q[1] = *(const uint4*)(p + 16);
  return f.v;
}
static __device__ __forceinline__ v8f wmma_bf16(v16bf a, v16bf b, v8f c) {
  return __builtin_amdgcn_wmma_f32_16x16x32_bf16(false, a, false, b, (short)0, c, false, false);
}

// ---------------- utility kernels ----------------
__global__ void k_f2bf(const float* __restrict__ s, unsigned short* __restrict__ d, int n) {
  for (int i = blockIdx.x * blockDim.x + threadIdx.x; i < n; i += gridDim.x * blockDim.x)
    d[i] = f2bf(s[i]);
}
__global__ void k_addvec(const float* __restrict__ a, const float* __restrict__ b,
                         float* __restrict__ c, int n) {
  for (int i = blockIdx.x * blockDim.x + threadIdx.x; i < n; i += gridDim.x * blockDim.x)
    c[i] = a[i] + b[i];
}
__global__ void k_zero(unsigned int* __restrict__ p, size_t n) {
  for (size_t i = blockIdx.x * (size_t)blockDim.x + threadIdx.x; i < n;
       i += (size_t)gridDim.x * blockDim.x)
    p[i] = 0u;
}

// ---------------- FiLM branch: Linear(512->512) + LN + SiLU + Linear(512->64) ----------------
__global__ void __launch_bounds__(256) k_film(const unsigned short* __restrict__ Ub,
    const unsigned short* __restrict__ w1, const float* __restrict__ b1,
    const float* __restrict__ lnw, const float* __restrict__ lnb,
    const unsigned short* __restrict__ w2, const float* __restrict__ b2,
    float* __restrict__ outp, int addOne) {
  __shared__ __align__(16) float lnt[16][HIDD];
  __shared__ __align__(16) unsigned short sil[16][HIDD];
  int lane = threadIdx.x & 31, wave = threadIdx.x >> 5;
  int row0 = blockIdx.x * 16;

  v8f c[4];
  for (int j = 0; j < 4; ++j) for (int r = 0; r < 8; ++r) c[j][r] = 0.0f;
  for (int k0 = 0; k0 < UDIM; k0 += 32) {
    v16bf a = load_a_frag(Ub, row0, UDIM, k0, lane);
    for (int j = 0; j < 4; ++j) {
      v16bf b = load_b_frag(w1, UDIM, (wave * 4 + j) * 16, k0, lane);
      c[j] = wmma_bf16(a, b, c[j]);
    }
  }
  int n = lane & 15, hif = lane >> 4;
  for (int j = 0; j < 4; ++j) {
    int col = (wave * 4 + j) * 16 + n;
    float bv = b1[col];
    for (int r = 0; r < 8; ++r) lnt[r + 8 * hif][col] = c[j][r] + bv;
  }
  __syncthreads();

  // LayerNorm + SiLU: wave handles rows 2w, 2w+1
  for (int rr = 0; rr < 2; ++rr) {
    int row = wave * 2 + rr;
    float s = 0.f, ss = 0.f;
    for (int cix = lane; cix < HIDD; cix += 32) { float v = lnt[row][cix]; s += v; ss += v * v; }
    for (int off = 16; off > 0; off >>= 1) { s += __shfl_xor(s, off, 32); ss += __shfl_xor(ss, off, 32); }
    float mu = s * (1.0f / HIDD);
    float var = ss * (1.0f / HIDD) - mu * mu;
    float rstd = rsqrtf(var + LN_EPS);
    for (int cix = lane; cix < HIDD; cix += 32) {
      float v = (lnt[row][cix] - mu) * rstd * lnw[cix] + lnb[cix];
      sil[row][cix] = f2bf(v * sigmoidf_(v));
    }
  }
  __syncthreads();

  if (wave < 4) {
    v8f c2; for (int r = 0; r < 8; ++r) c2[r] = 0.0f;
    for (int k0 = 0; k0 < HIDD; k0 += 32) {
      v16bf a = load_a_frag(&sil[0][0], 0, HIDD, k0, lane);
      v16bf b = load_b_frag(w2, HIDD, wave * 16, k0, lane);
      c2 = wmma_bf16(a, b, c2);
    }
    int col = wave * 16 + n;
    float bv = b2[col] + (addOne ? 1.0f : 0.0f);
    for (int r = 0; r < 8; ++r) {
      int brow = row0 + r + 8 * hif;
      outp[(size_t)brow * EDIM + col] = c2[r] + bv;
    }
  }
}

// ---------------- e0 = U @ u0_w.T + u0_b, written as bf16 into xs[0] ----------------
__global__ void __launch_bounds__(128) k_e0(const unsigned short* __restrict__ Ub,
    const unsigned short* __restrict__ w, const float* __restrict__ bias,
    unsigned short* __restrict__ xs0) {
  int lane = threadIdx.x & 31, wave = threadIdx.x >> 5;
  int row0 = blockIdx.x * 16;
  v8f c; for (int r = 0; r < 8; ++r) c[r] = 0.0f;
  for (int k0 = 0; k0 < UDIM; k0 += 32) {
    v16bf a = load_a_frag(Ub, row0, UDIM, k0, lane);
    v16bf b = load_b_frag(w, UDIM, wave * 16, k0, lane);
    c = wmma_bf16(a, b, c);
  }
  int n = lane & 15, hif = lane >> 4;
  int col = wave * 16 + n;
  float bv = bias[col];
  for (int r = 0; r < 8; ++r) {
    int brow = row0 + r + 8 * hif;
    xs0[(size_t)brow * EDIM + col] = f2bf(c[r] + bv);
  }
}

// ---------------- xs[t>=1] = bf16(emb[X[:,t-1]] * (1+gamma) + beta) ----------------
__global__ void k_build_xs(const float* __restrict__ emb, const int* __restrict__ X,
                           const float* __restrict__ g1, const float* __restrict__ bt,
                           unsigned short* __restrict__ xs) {
  size_t total = (size_t)(DDIM - 1) * BDIM * EDIM;
  for (size_t g = blockIdx.x * (size_t)blockDim.x + threadIdx.x; g < total;
       g += (size_t)gridDim.x * blockDim.x) {
    int t = 1 + (int)(g / ((size_t)BDIM * EDIM));
    int rem = (int)(g % ((size_t)BDIM * EDIM));
    int b = rem / EDIM, e = rem % EDIM;
    int tok = X[(size_t)b * DDIM + (t - 1)];
    float v = emb[tok * EDIM + e] * g1[(size_t)b * EDIM + e] + bt[(size_t)b * EDIM + e];
    xs[((size_t)t * BDIM + b) * EDIM + e] = f2bf(v);
  }
}

// ---------------- LSTM layer 0 step: gates = xs_t@wih.T + h@whh.T + bsum ----------------
__global__ void __launch_bounds__(256) k_lstm0(const unsigned short* __restrict__ xst,
    const unsigned short* __restrict__ wih, const unsigned short* __restrict__ whh,
    const float* __restrict__ bsum, unsigned short* __restrict__ hb, float* __restrict__ cst) {
  __shared__ __align__(16) unsigned short xpan[16][EDIM];   // 2 KiB
  __shared__ __align__(16) unsigned short hpan[16][HDIM];   // 8 KiB
  int lane = threadIdx.x & 31, wave = threadIdx.x >> 5, tid = threadIdx.x;
  int row0 = blockIdx.x * 16;

  // async-stage the block's contiguous A panels (x: 16x64, h: 16x256) into LDS
  {
    const char* xg = (const char*)(xst + (size_t)row0 * EDIM);
    const char* hg = (const char*)(hb + (size_t)row0 * HDIM);
    unsigned xl = lds_off(&xpan[0][0]);
    unsigned hl = lds_off(&hpan[0][0]);
    const int XCH = 16 * EDIM * 2 / 16;              // 128 chunks
    const int HCH = 16 * HDIM * 2 / 16;              // 512 chunks
    for (int i = tid; i < XCH + HCH; i += 256) {
      if (i < XCH) async_copy16(xl + i * 16, xg + i * 16);
      else { int j = i - XCH; async_copy16(hl + j * 16, hg + j * 16); }
    }
    async_wait0();
    __syncthreads();
  }

  v8f c[8];
  for (int q = 0; q < 8; ++q) for (int r = 0; r < 8; ++r) c[q][r] = 0.0f;
  int n = lane & 15, hif = lane >> 4;
  for (int k0 = 0; k0 < EDIM; k0 += 32) {
    v16bf a = load_a_frag(&xpan[0][0], 0, EDIM, k0, lane);
    for (int q = 0; q < 8; ++q) {
      int n0 = (wave + 8 * q) * 16;
      v16bf b = load_b_frag(wih, EDIM, n0, k0, lane);
      if (k0 + 32 < EDIM)
        __builtin_prefetch(wih + (size_t)(n0 + n) * EDIM + k0 + 32, 0, 3);
      c[q] = wmma_bf16(a, b, c[q]);
    }
  }
  for (int k0 = 0; k0 < HDIM; k0 += 32) {
    v16bf a = load_a_frag(&hpan[0][0], 0, HDIM, k0, lane);
    for (int q = 0; q < 8; ++q) {
      int n0 = (wave + 8 * q) * 16;
      v16bf b = load_b_frag(whh, HDIM, n0, k0, lane);
      if (k0 + 32 < HDIM)
        __builtin_prefetch(whh + (size_t)(n0 + n) * HDIM + k0 + 32, 0, 3);
      c[q] = wmma_bf16(a, b, c[q]);
    }
  }
  for (int q = 0; q < 8; ++q) {
    float bv = bsum[(wave + 8 * q) * 16 + n];
    for (int r = 0; r < 8; ++r) c[q][r] += bv;
  }
  // gates for column j live in this wave's registers: i=c[s], f=c[2+s], g=c[4+s], o=c[6+s]
  for (int s = 0; s < 2; ++s) {
    int j = 16 * wave + n + 128 * s;
    for (int r = 0; r < 8; ++r) {
      int brow = row0 + r + 8 * hif;
      size_t idx = (size_t)brow * HDIM + j;
      float ig = sigmoidf_(c[0 + s][r]);
      float fg = sigmoidf_(c[2 + s][r]);
      float gg = tanhf(c[4 + s][r]);
      float og = sigmoidf_(c[6 + s][r]);
      float cn = fg * cst[idx] + ig * gg;
      float hn = og * tanhf(cn);
      cst[idx] = cn;
      hb[idx] = f2bf(hn);
    }
  }
}

// ---------------- LSTM layer 1 step + fused head (logits, log-softmax, gather, accumulate) ---
__global__ void __launch_bounds__(256) k_lstm1(int t, const unsigned short* __restrict__ h0b,
    const unsigned short* __restrict__ wih, const unsigned short* __restrict__ whh,
    const float* __restrict__ bsum, unsigned short* __restrict__ h1b, float* __restrict__ c1,
    const unsigned short* __restrict__ hwb, const float* __restrict__ hbias,
    const int* __restrict__ X, float* __restrict__ out) {
  __shared__ __align__(16) unsigned short p0[16][HDIM];  // 8 KiB (layer0 h panel)
  __shared__ __align__(16) unsigned short p1[16][HDIM];  // 8 KiB (layer1 h panel)
  __shared__ __align__(16) float ht[16][HDIM];           // 16 KiB (new h, f32, for head)
  __shared__ float lg[16][KDIM];
  int lane = threadIdx.x & 31, wave = threadIdx.x >> 5, tid = threadIdx.x;
  int row0 = blockIdx.x * 16;

  {
    const char* g0 = (const char*)(h0b + (size_t)row0 * HDIM);
    const char* g1 = (const char*)(h1b + (size_t)row0 * HDIM);
    unsigned l0 = lds_off(&p0[0][0]);
    unsigned l1 = lds_off(&p1[0][0]);
    const int HCH = 16 * HDIM * 2 / 16;  // 512 chunks each
    for (int i = tid; i < 2 * HCH; i += 256) {
      if (i < HCH) async_copy16(l0 + i * 16, g0 + i * 16);
      else { int j = i - HCH; async_copy16(l1 + j * 16, g1 + j * 16); }
    }
    async_wait0();
    __syncthreads();
  }

  v8f c[8];
  for (int q = 0; q < 8; ++q) for (int r = 0; r < 8; ++r) c[q][r] = 0.0f;
  int n = lane & 15, hif = lane >> 4;
  for (int k0 = 0; k0 < HDIM; k0 += 32) {
    v16bf a = load_a_frag(&p0[0][0], 0, HDIM, k0, lane);
    for (int q = 0; q < 8; ++q) {
      int n0 = (wave + 8 * q) * 16;
      v16bf b = load_b_frag(wih, HDIM, n0, k0, lane);
      if (k0 + 32 < HDIM)
        __builtin_prefetch(wih + (size_t)(n0 + n) * HDIM + k0 + 32, 0, 3);
      c[q] = wmma_bf16(a, b, c[q]);
    }
  }
  for (int k0 = 0; k0 < HDIM; k0 += 32) {
    v16bf a = load_a_frag(&p1[0][0], 0, HDIM, k0, lane);
    for (int q = 0; q < 8; ++q) {
      int n0 = (wave + 8 * q) * 16;
      v16bf b = load_b_frag(whh, HDIM, n0, k0, lane);
      if (k0 + 32 < HDIM)
        __builtin_prefetch(whh + (size_t)(n0 + n) * HDIM + k0 + 32, 0, 3);
      c[q] = wmma_bf16(a, b, c[q]);
    }
  }
  for (int q = 0; q < 8; ++q) {
    float bv = bsum[(wave + 8 * q) * 16 + n];
    for (int r = 0; r < 8; ++r) c[q][r] += bv;
  }
  for (int s = 0; s < 2; ++s) {
    int j = 16 * wave + n + 128 * s;
    for (int r = 0; r < 8; ++r) {
      int brow = row0 + r + 8 * hif;
      int m = r + 8 * hif;
      size_t idx = (size_t)brow * HDIM + j;
      float ig = sigmoidf_(c[0 + s][r]);
      float fg = sigmoidf_(c[2 + s][r]);
      float gg = tanhf(c[4 + s][r]);
      float og = sigmoidf_(c[6 + s][r]);
      float cn = fg * c1[idx] + ig * gg;
      float hn = og * tanhf(cn);
      c1[idx] = cn;
      h1b[idx] = f2bf(hn);
      ht[m][j] = hn;
    }
  }
  __syncthreads();
  // head: logits[16][20] = ht @ head_w.T + head_b
  for (int p = tid; p < 16 * KDIM; p += 256) {
    int row = p / KDIM, kk = p % KDIM;
    float acc = hbias[kk];
    const unsigned short* wr = hwb + (size_t)kk * HDIM;
    for (int x = 0; x < HDIM; ++x) acc += ht[row][x] * bf2f(wr[x]);
    lg[row][kk] = acc;
  }
  __syncthreads();
  if (tid < 16) {
    int brow = row0 + tid;
    float mx = -1e30f;
    for (int kk = 0; kk < KDIM; ++kk) mx = fmaxf(mx, lg[tid][kk]);
    float se = 0.f;
    for (int kk = 0; kk < KDIM; ++kk) se += expf(lg[tid][kk] - mx);
    int tok = X[(size_t)brow * DDIM + t];
    out[brow] += (lg[tid][tok] - mx) - logf(se);
  }
}

// ---------------- host driver ----------------
extern "C" void kernel_launch(void* const* d_in, const int* in_sizes, int n_in,
                              void* d_out, int out_size, void* d_ws, size_t ws_size,
                              hipStream_t stream) {
  (void)in_sizes; (void)n_in; (void)out_size; (void)ws_size;
  const float* U    = (const float*)d_in[0];
  const int*   X    = (const int*)d_in[1];
  const float* emb  = (const float*)d_in[2];
  const float* u0_w = (const float*)d_in[3];
  const float* u0_b = (const float*)d_in[4];
  const float* gw1  = (const float*)d_in[5];
  const float* gb1  = (const float*)d_in[6];
  const float* glnw = (const float*)d_in[7];
  const float* glnb = (const float*)d_in[8];
  const float* gw2  = (const float*)d_in[9];
  const float* gb2  = (const float*)d_in[10];
  const float* bw1  = (const float*)d_in[11];
  const float* bb1  = (const float*)d_in[12];
  const float* blnw = (const float*)d_in[13];
  const float* blnb = (const float*)d_in[14];
  const float* bw2  = (const float*)d_in[15];
  const float* bb2  = (const float*)d_in[16];
  const float* wih0 = (const float*)d_in[17];
  const float* whh0 = (const float*)d_in[18];
  const float* bih0 = (const float*)d_in[19];
  const float* bhh0 = (const float*)d_in[20];
  const float* wih1 = (const float*)d_in[21];
  const float* whh1 = (const float*)d_in[22];
  const float* bih1 = (const float*)d_in[23];
  const float* bhh1 = (const float*)d_in[24];
  const float* head_w = (const float*)d_in[25];
  const float* head_b = (const float*)d_in[26];
  float* out = (float*)d_out;

  char* ws = (char*)d_ws;
  size_t off = 0;
  auto alloc = [&](size_t bytes) -> char* {
    char* p = ws + off;
    off = (off + bytes + 255) & ~(size_t)255;
    return p;
  };
  unsigned short* Ub    = (unsigned short*)alloc((size_t)BDIM * UDIM * 2);
  unsigned short* gw1b  = (unsigned short*)alloc((size_t)HIDD * UDIM * 2);
  unsigned short* bw1b  = (unsigned short*)alloc((size_t)HIDD * UDIM * 2);
  unsigned short* gw2b  = (unsigned short*)alloc((size_t)EDIM * HIDD * 2);
  unsigned short* bw2b  = (unsigned short*)alloc((size_t)EDIM * HIDD * 2);
  unsigned short* u0wb  = (unsigned short*)alloc((size_t)EDIM * UDIM * 2);
  unsigned short* wih0b = (unsigned short*)alloc((size_t)G4H * EDIM * 2);
  unsigned short* whh0b = (unsigned short*)alloc((size_t)G4H * HDIM * 2);
  unsigned short* wih1b = (unsigned short*)alloc((size_t)G4H * HDIM * 2);
  unsigned short* whh1b = (unsigned short*)alloc((size_t)G4H * HDIM * 2);
  unsigned short* hwb   = (unsigned short*)alloc((size_t)KDIM * HDIM * 2);
  float* gam1  = (float*)alloc((size_t)BDIM * EDIM * 4);
  float* bet   = (float*)alloc((size_t)BDIM * EDIM * 4);
  float* bsum0 = (float*)alloc((size_t)G4H * 4);
  float* bsum1 = (float*)alloc((size_t)G4H * 4);
  unsigned short* xs = (unsigned short*)alloc((size_t)DDIM * BDIM * EDIM * 2);
  // state (contiguous for single zero pass): h0b, h1b (bf16), c0, c1 (f32)
  unsigned short* h0b = (unsigned short*)alloc((size_t)BDIM * HDIM * 2);
  unsigned short* h1b = (unsigned short*)alloc((size_t)BDIM * HDIM * 2);
  float* c0 = (float*)alloc((size_t)BDIM * HDIM * 4);
  float* c1 = (float*)alloc((size_t)BDIM * HDIM * 4);

  auto conv = [&](const float* s, unsigned short* d, int n) {
    k_f2bf<<<512, 256, 0, stream>>>(s, d, n);
  };
  conv(U, Ub, BDIM * UDIM);
  conv(gw1, gw1b, HIDD * UDIM);
  conv(bw1, bw1b, HIDD * UDIM);
  conv(gw2, gw2b, EDIM * HIDD);
  conv(bw2, bw2b, EDIM * HIDD);
  conv(u0_w, u0wb, EDIM * UDIM);
  conv(wih0, wih0b, G4H * EDIM);
  conv(whh0, whh0b, G4H * HDIM);
  conv(wih1, wih1b, G4H * HDIM);
  conv(whh1, whh1b, G4H * HDIM);
  conv(head_w, hwb, KDIM * HDIM);

  k_addvec<<<4, 256, 0, stream>>>(bih0, bhh0, bsum0, G4H);
  k_addvec<<<4, 256, 0, stream>>>(bih1, bhh1, bsum1, G4H);

  // zero h0b..c1 region (bf16 sizes are 256-aligned so the region is contiguous) and out
  size_t state_u32 = ((size_t)BDIM * HDIM * 2 * 2 + (size_t)BDIM * HDIM * 4 * 2) / 4;
  k_zero<<<512, 256, 0, stream>>>((unsigned int*)h0b, state_u32);
  k_zero<<<8, 256, 0, stream>>>((unsigned int*)out, (size_t)BDIM);

  // FiLM branches (gamma writes 1+gamma), e0 into xs[0], modulated embeddings into xs[1..]
  k_film<<<BDIM / 16, 256, 0, stream>>>(Ub, gw1b, gb1, glnw, glnb, gw2b, gb2, gam1, 1);
  k_film<<<BDIM / 16, 256, 0, stream>>>(Ub, bw1b, bb1, blnw, blnb, bw2b, bb2, bet, 0);
  k_e0<<<BDIM / 16, 128, 0, stream>>>(Ub, u0wb, u0_b, xs);
  k_build_xs<<<4096, 256, 0, stream>>>(emb, X, gam1, bet, xs);

  // sequential recurrence: stream order carries the t -> t+1 and layer0 -> layer1 deps
  for (int t = 0; t < DDIM; ++t) {
    const unsigned short* xst = xs + (size_t)t * BDIM * EDIM;
    k_lstm0<<<BDIM / 16, 256, 0, stream>>>(xst, wih0b, whh0b, bsum0, h0b, c0);
    k_lstm1<<<BDIM / 16, 256, 0, stream>>>(t, h0b, wih1b, whh1b, bsum1, h1b, c1,
                                           hwb, head_b, X, out);
  }
}
